// LSTM_58514634440709
// MI455X (gfx1250) — compile-verified, matrix operand order checked by reading, not run on hardware
//
#include <hip/hip_runtime.h>

typedef _Float16 v16h __attribute__((ext_vector_type(16)));
typedef _Float16 h8   __attribute__((ext_vector_type(8)));
typedef _Float16 h2   __attribute__((ext_vector_type(2)));
typedef float    v8f  __attribute__((ext_vector_type(8)));

#define HD    96
#define G4    384
#define BATCH 8
#define SEQ   32768
#define NWAVE 12
#define TPW   2     // N-tiles per wave; each A-fragment is reused TPW times
#define NTHR  384   // 12 waves; update phase: each thread owns states (b,j),(b,j+1)

#if __has_builtin(__builtin_amdgcn_global_load_async_to_lds_b128)
#define XS_ASYNC 1
typedef int v4i_t __attribute__((ext_vector_type(4)));
typedef __attribute__((address_space(1))) v4i_t g_v4i;   // global-AS int4 (builtin src type)
typedef __attribute__((address_space(3))) v4i_t l_v4i;   // LDS-AS int4 (builtin dst type)
#else
#define XS_ASYNC 0
#endif

// Workgroup barrier that only drains the LDS counter (DScnt).
// __syncthreads() waits LOADcnt/STOREcnt too, which would serialize the
// in-flight global x traffic / out stores into every barrier. Only LDS data
// (gates, h-state) must be ordered across these barriers; the async x refill
// is ordered explicitly with s_wait_asynccnt at refill steps.
__device__ __forceinline__ void wg_barrier_lds() {
    asm volatile(
        "s_wait_dscnt 0x0\n\t"
        "s_barrier_signal -1\n\t"
        "s_barrier_wait -1"
        ::: "memory");
}
__device__ __forceinline__ void wait_async0() {
    asm volatile("s_wait_asynccnt 0x0" ::: "memory");
}

// ---- fast activations: prefer gfx1250 v_tanh_f32; fall back to exp + fast rcp ----
__device__ __forceinline__ float fast_rcp(float x) {
#if __has_builtin(__builtin_amdgcn_rcpf)
    return __builtin_amdgcn_rcpf(x);
#else
    return 1.0f / x;
#endif
}
__device__ __forceinline__ float tanh_(float x) {
#if __has_builtin(__builtin_amdgcn_tanhf)
    return __builtin_amdgcn_tanhf(x);
#else
    x = fminf(15.0f, fmaxf(-15.0f, x));
    float e = __expf(-2.0f * x);
    return (1.0f - e) * fast_rcp(1.0f + e);
#endif
}
__device__ __forceinline__ float sig_(float x) {
#if __has_builtin(__builtin_amdgcn_tanhf)
    return fmaf(0.5f, __builtin_amdgcn_tanhf(0.5f * x), 0.5f);
#else
    return fast_rcp(1.0f + __expf(-x));
#endif
}

// Load one 16x32 f16 A-fragment from LDS, per documented CDNA5 layout:
// lanes 0-15: row M=lane, K = k0+{0..7} (halves 0-7) and k0+{16..23} (halves 8-15)
// lanes 16-31: row M=lane-16, K = k0+{8..15} and k0+{24..31}
__device__ __forceinline__ v16h load_a_frag(const _Float16* base, int pitch, int lane, int k0) {
    int row = lane & 15;
    int kb  = k0 + ((lane >> 4) << 3);
    const _Float16* p = base + row * pitch + kb;
    h8 lo = *(const h8*)(p);        // ds_read_b128
    h8 hi = *(const h8*)(p + 16);   // ds_read_b128
    v16h a;
#pragma unroll
    for (int i = 0; i < 8; ++i) { a[i] = lo[i]; a[i + 8] = hi[i]; }
    return a;
}

__global__ __launch_bounds__(NTHR, 1)
void lstm_fused_kernel(const float* __restrict__ x,      // (8, 32768)
                       const float* __restrict__ wih0,   // (384, 1)
                       const float* __restrict__ whh0,   // (384, 96)
                       const float* __restrict__ b0,     // (384,)
                       const float* __restrict__ wih1,   // (384, 96)
                       const float* __restrict__ whh1,   // (384, 96)
                       const float* __restrict__ b1,     // (384,)
                       const float* __restrict__ h0in,   // (2, 96)
                       const float* __restrict__ c0in,   // (2, 96)
                       const float* __restrict__ headw,  // (96,)
                       const float* __restrict__ headb,  // (1,)
                       float* __restrict__ out)          // (8, 32768)
{
    __shared__ __align__(16) _Float16 A0[16 * HD];        // h0_prev  (rows 8..15 zero)
    __shared__ __align__(16) _Float16 A1[16 * 2 * HD];    // [h0_cur | h1_prev]
    __shared__ float G0[BATCH * G4];                      // layer-0 WMMA gates
    __shared__ float G1[BATCH * G4];                      // layer-1 WMMA gates
    __shared__ float red[2 * NWAVE];                      // head partials (one per half-wave)
    __shared__ __align__(16) float xs8[2][BATCH * 8];     // x, double-buffered 8-step groups

    const int tid  = threadIdx.x;
    const int lane = tid & 31;
    const int wave = tid >> 5;          // owns N-tiles 2*wave, 2*wave+1

    // ---- init LDS A buffers from initial states --------------------------------
    for (int idx = tid; idx < 16 * HD; idx += NTHR) {
        int r = idx / HD, c = idx % HD;
        A0[idx] = (r < BATCH) ? (_Float16)h0in[c] : (_Float16)0.0f;
    }
    for (int idx = tid; idx < 16 * 2 * HD; idx += NTHR) {
        int r = idx / (2 * HD), c = idx % (2 * HD);
        _Float16 v = (_Float16)0.0f;
        if (r < BATCH && c >= HD) v = (_Float16)h0in[HD + (c - HD)];
        A1[idx] = v;
    }

    // ---- per-wave register-resident B fragments (weights, f32 -> f16 once) -----
    const int khiB = (lane >> 4) << 3;
    v16h b0f[TPW][3], b1f[TPW][6];
#pragma unroll
    for (int tt = 0; tt < TPW; ++tt) {
        const int colB = (TPW * wave + tt) * 16 + (lane & 15);
#pragma unroll
        for (int f = 0; f < 3; ++f) {
            int k0 = f * 32 + khiB;
#pragma unroll
            for (int i = 0; i < 16; ++i) {
                int K = k0 + (i < 8 ? i : 8 + i);
                b0f[tt][f][i] = (_Float16)whh0[colB * HD + K];
            }
        }
#pragma unroll
        for (int f = 0; f < 6; ++f) {
            int k0 = f * 32 + khiB;
#pragma unroll
            for (int i = 0; i < 16; ++i) {
                int K = k0 + (i < 8 ? i : 8 + i);      // 0..191: [W_ih_1 | W_hh_1]
                float w = (K < HD) ? wih1[colB * HD + K] : whh1[colB * HD + (K - HD)];
                b1f[tt][f][i] = (_Float16)w;
            }
        }
    }

    // ---- per-thread state: two adjacent hidden units of one batch --------------
    const int upIdx = tid * 2;          // states (b, j) and (b, j+1)
    const int b = upIdx / HD;
    const int j = upIdx % HD;
    float wi_i0 = wih0[j],        wi_i1 = wih0[j + 1];
    float wi_f0 = wih0[HD + j],   wi_f1 = wih0[HD + j + 1];
    float wi_g0 = wih0[2*HD + j], wi_g1 = wih0[2*HD + j + 1];
    float wi_o0 = wih0[3*HD + j], wi_o1 = wih0[3*HD + j + 1];
    float b0_i0 = b0[j],        b0_i1 = b0[j + 1];
    float b0_f0 = b0[HD + j],   b0_f1 = b0[HD + j + 1];
    float b0_g0 = b0[2*HD + j], b0_g1 = b0[2*HD + j + 1];
    float b0_o0 = b0[3*HD + j], b0_o1 = b0[3*HD + j + 1];
    float b1_i0 = b1[j],        b1_i1 = b1[j + 1];
    float b1_f0 = b1[HD + j],   b1_f1 = b1[HD + j + 1];
    float b1_g0 = b1[2*HD + j], b1_g1 = b1[2*HD + j + 1];
    float b1_o0 = b1[3*HD + j], b1_o1 = b1[3*HD + j + 1];
    const float hw0 = headw[j], hw1 = headw[j + 1];
    const float hb = headb[0];
    float c0r0 = c0in[j],      c0r1 = c0in[j + 1];
    float c1r0 = c0in[HD + j], c1r1 = c0in[HD + j + 1];

#if XS_ASYNC
    // Prologue: async-copy x group 0 into xs8[0] (ASYNCcnt-tracked; never
    // drained by wg_barrier_lds, only by explicit s_wait_asynccnt at refills).
    if (tid < 2 * BATCH) {
        int bb = tid >> 1, hh = tid & 1;
        __builtin_amdgcn_global_load_async_to_lds_b128(
            (g_v4i*)(x + bb * SEQ + hh * 4),
            (l_v4i*)(&xs8[0][bb * 8 + hh * 4]), 0, 0);
    }
#else
    float4 xqa0, xqa1, xqb0, xqb1;   // current group (8 vals) + next group in flight
    if (tid < BATCH) {
        xqa0 = *(const float4*)(x + tid * SEQ);
        xqa1 = *(const float4*)(x + tid * SEQ + 4);
        xqb0 = *(const float4*)(x + tid * SEQ + 8);
        xqb1 = *(const float4*)(x + tid * SEQ + 12);
    }
#endif

    __syncthreads();

#pragma unroll 1
    for (int t = 0; t < SEQ; ++t) {
        // --- s1: layer-0 WMMA: gates0 = h0_prev @ W_hh_0^T (A-frag reused x2) ---
        {
            v16h a0 = load_a_frag(A0, HD, lane, 0);
            v16h a1 = load_a_frag(A0, HD, lane, 32);
            v16h a2 = load_a_frag(A0, HD, lane, 64);
            v8f acc0 = {}, acc1 = {};
            acc0 = __builtin_amdgcn_wmma_f32_16x16x32_f16(false, a0, false, b0f[0][0], (short)0, acc0, false, false);
            acc1 = __builtin_amdgcn_wmma_f32_16x16x32_f16(false, a0, false, b0f[1][0], (short)0, acc1, false, false);
            acc0 = __builtin_amdgcn_wmma_f32_16x16x32_f16(false, a1, false, b0f[0][1], (short)0, acc0, false, false);
            acc1 = __builtin_amdgcn_wmma_f32_16x16x32_f16(false, a1, false, b0f[1][1], (short)0, acc1, false, false);
            acc0 = __builtin_amdgcn_wmma_f32_16x16x32_f16(false, a2, false, b0f[0][2], (short)0, acc0, false, false);
            acc1 = __builtin_amdgcn_wmma_f32_16x16x32_f16(false, a2, false, b0f[1][2], (short)0, acc1, false, false);
            if (lane < 16) {
                int c0c = (TPW * wave + 0) * 16 + lane;
                int c1c = (TPW * wave + 1) * 16 + lane;
#pragma unroll
                for (int r = 0; r < BATCH; ++r) { G0[r * G4 + c0c] = acc0[r]; G0[r * G4 + c1c] = acc1[r]; }
            }
        }
        // --- x refill: once per 8 steps ----------------------------------------
#if XS_ASYNC
        if ((t & 7) == 0 && tid < 2 * BATCH) {
            wait_async0();                                 // group-t copy (issued t-8) done
            int tn = t + 8; if (tn > SEQ - 8) tn = SEQ - 8;  // clamped; tail garbage unused
            int bb = tid >> 1, hh = tid & 1;
            __builtin_amdgcn_global_load_async_to_lds_b128(
                (g_v4i*)(x + bb * SEQ + tn + hh * 4),
                (l_v4i*)(&xs8[((t >> 3) & 1) ^ 1][bb * 8 + hh * 4]), 0, 0);
        }
#else
        if ((t & 7) == 0 && tid < BATCH) {
            *(float4*)(&xs8[(t >> 3) & 1][tid * 8])     = xqa0;
            *(float4*)(&xs8[(t >> 3) & 1][tid * 8 + 4]) = xqa1;
            xqa0 = xqb0; xqa1 = xqb1;
            int tn = t + 16; if (tn > SEQ - 8) tn = SEQ - 8;
            xqb0 = *(const float4*)(x + tid * SEQ + tn);
            xqb1 = *(const float4*)(x + tid * SEQ + tn + 4);
        }
#endif
        wg_barrier_lds();   // B1: G0 + xs8 ready; red(t-1) safe to read

        // --- s2: head output for t-1, then layer-0 update -----------------------
        if (t > 0 && tid < BATCH)
            out[tid * SEQ + (t - 1)] = red[3 * tid] + red[3 * tid + 1] + red[3 * tid + 2] + hb;
        {
            float xv = xs8[(t >> 3) & 1][b * 8 + (t & 7)];
            float gi0 = G0[b * G4 + j]          + xv * wi_i0 + b0_i0;
            float gi1 = G0[b * G4 + j + 1]      + xv * wi_i1 + b0_i1;
            float gf0 = G0[b * G4 + HD + j]     + xv * wi_f0 + b0_f0;
            float gf1 = G0[b * G4 + HD + j + 1] + xv * wi_f1 + b0_f1;
            float gg0 = G0[b * G4 + 2*HD + j]     + xv * wi_g0 + b0_g0;
            float gg1 = G0[b * G4 + 2*HD + j + 1] + xv * wi_g1 + b0_g1;
            float go0 = G0[b * G4 + 3*HD + j]     + xv * wi_o0 + b0_o0;
            float go1 = G0[b * G4 + 3*HD + j + 1] + xv * wi_o1 + b0_o1;
            c0r0 = sig_(gf0) * c0r0 + sig_(gi0) * tanh_(gg0);
            c0r1 = sig_(gf1) * c0r1 + sig_(gi1) * tanh_(gg1);
            float h0v0 = sig_(go0) * tanh_(c0r0);
            float h0v1 = sig_(go1) * tanh_(c0r1);
            h2 hp; hp[0] = (_Float16)h0v0; hp[1] = (_Float16)h0v1;
            *(h2*)(&A0[b * HD + j]) = hp;              // packed b32 store
            *(h2*)(&A1[b * 2 * HD + j]) = hp;
        }
        wg_barrier_lds();   // B2: h0 ready for layer-1 matmul

        // --- s3: layer-1 WMMA: gates1 = [h0_t | h1_prev] @ [W_ih_1;W_hh_1]^T ----
        {
            v8f acc0 = {}, acc1 = {};
            v16h a = load_a_frag(A1, 2 * HD, lane, 0);
#pragma unroll
            for (int f = 0; f < 6; ++f) {
                v16h an;
                if (f < 5) an = load_a_frag(A1, 2 * HD, lane, (f + 1) * 32); // prefetch next frag
                acc0 = __builtin_amdgcn_wmma_f32_16x16x32_f16(false, a, false, b1f[0][f], (short)0, acc0, false, false);
                acc1 = __builtin_amdgcn_wmma_f32_16x16x32_f16(false, a, false, b1f[1][f], (short)0, acc1, false, false);
                a = an;
            }
            if (lane < 16) {
                int c0c = (TPW * wave + 0) * 16 + lane;
                int c1c = (TPW * wave + 1) * 16 + lane;
#pragma unroll
                for (int r = 0; r < BATCH; ++r) { G1[r * G4 + c0c] = acc0[r]; G1[r * G4 + c1c] = acc1[r]; }
            }
        }
        wg_barrier_lds();   // B3: G1 ready

        // --- s4: layer-1 update + head partial (no trailing barrier) ------------
        {
            float gi0 = G1[b * G4 + j],          gi1 = G1[b * G4 + j + 1];
            float gf0 = G1[b * G4 + HD + j],     gf1 = G1[b * G4 + HD + j + 1];
            float gg0 = G1[b * G4 + 2*HD + j],   gg1 = G1[b * G4 + 2*HD + j + 1];
            float go0 = G1[b * G4 + 3*HD + j],   go1 = G1[b * G4 + 3*HD + j + 1];
            c1r0 = sig_(gf0 + b1_f0) * c1r0 + sig_(gi0 + b1_i0) * tanh_(gg0 + b1_g0);
            c1r1 = sig_(gf1 + b1_f1) * c1r1 + sig_(gi1 + b1_i1) * tanh_(gg1 + b1_g1);
            float h1v0 = sig_(go0 + b1_o0) * tanh_(c1r0);
            float h1v1 = sig_(go1 + b1_o1) * tanh_(c1r1);
            h2 hp; hp[0] = (_Float16)h1v0; hp[1] = (_Float16)h1v1;
            *(h2*)(&A1[b * 2 * HD + HD + j]) = hp;     // h1_prev for next step
            // head partial: both states share batch b; half-wave (16-lane) tree
            float v = hw0 * h1v0 + hw1 * h1v1;
            v += __shfl_xor(v, 1, 32);
            v += __shfl_xor(v, 2, 32);
            v += __shfl_xor(v, 4, 32);
            v += __shfl_xor(v, 8, 32);
            if ((lane & 15) == 0) red[2 * wave + (lane >> 4)] = v;  // half-wave h -> batch h/3
        }
        // no barrier here: next-step B1 orders red/G1/A1 reuse
    }

    wg_barrier_lds();
    if (tid < BATCH)
        out[tid * SEQ + (SEQ - 1)] = red[3 * tid] + red[3 * tid + 1] + red[3 * tid + 2] + hb;
}

extern "C" void kernel_launch(void* const* d_in, const int* in_sizes, int n_in,
                              void* d_out, int out_size, void* d_ws, size_t ws_size,
                              hipStream_t stream) {
    (void)in_sizes; (void)n_in; (void)out_size; (void)d_ws; (void)ws_size;
    const float* x     = (const float*)d_in[0];
    const float* wih0  = (const float*)d_in[1];
    const float* whh0  = (const float*)d_in[2];
    const float* b0    = (const float*)d_in[3];
    const float* wih1  = (const float*)d_in[4];
    const float* whh1  = (const float*)d_in[5];
    const float* b1    = (const float*)d_in[6];
    const float* h0in  = (const float*)d_in[7];
    const float* c0in  = (const float*)d_in[8];
    const float* headw = (const float*)d_in[9];
    const float* headb = (const float*)d_in[10];
    float* out = (float*)d_out;

    hipLaunchKernelGGL(lstm_fused_kernel, dim3(1), dim3(NTHR), 0, stream,
                       x, wih0, whh0, b0, wih1, whh1, b1, h0in, c0in, headw, headb, out);
}
